// CausalSelfAttention_72791105732727
// MI455X (gfx1250) — matrix operand order repaired and checked by reading, not compiled, not run on hardware
//
#include <hip/hip_runtime.h>
#include <hip/hip_bf16.h>

// ---------------------------------------------------------------------------
// CDNA5 (gfx1250) causal self-attention: bf16 WMMA + async global->LDS copies.
// Pipeline: convert/transposes -> QKV GEMM (wmma, async double-buffer)
//           -> V transpose -> flash attention (wmma, async double-buffer,
//              diagonal tile peeled) -> output projection GEMM (wmma)
// ---------------------------------------------------------------------------

typedef __attribute__((ext_vector_type(16))) __bf16 v16bf;
typedef __attribute__((ext_vector_type(8)))  float  v8f;

union Frag16 { uint4 u[2]; v16bf v; };

// A-fragment: one contiguous 32B run (K = kb..kb+15)
static __device__ __forceinline__ v16bf load_frag_halves(const __bf16* p0, const __bf16* p1) {
  Frag16 f;
  (void)p1;
  f.u[0] = *reinterpret_cast<const uint4*>(p0);
  f.u[1] = *reinterpret_cast<const uint4*>(p0 + 8);
  return f.v;
}
// B-fragment: one contiguous 32B run (K = kh..kh+15)
static __device__ __forceinline__ v16bf load_frag_contig(const __bf16* p) {
  Frag16 f;
  const uint4* q = reinterpret_cast<const uint4*>(p);
  f.u[0] = q[0];
  f.u[1] = q[1];
  return f.v;
}

static __device__ __forceinline__ float half16_max(float x) {
  x = fmaxf(x, __shfl_xor(x, 1, 32));
  x = fmaxf(x, __shfl_xor(x, 2, 32));
  x = fmaxf(x, __shfl_xor(x, 4, 32));
  x = fmaxf(x, __shfl_xor(x, 8, 32));
  return x;
}
static __device__ __forceinline__ float half16_sum(float x) {
  x += __shfl_xor(x, 1, 32);
  x += __shfl_xor(x, 2, 32);
  x += __shfl_xor(x, 4, 32);
  x += __shfl_xor(x, 8, 32);
  return x;
}

// ---------------------------------------------------------------------------
// CDNA5 async global->LDS (tracked by ASYNCcnt). GVS addressing:
//   mem = SGPR64 + VGPR32_offset ; LDS dest = per-lane VGPR LDS byte address.
// ---------------------------------------------------------------------------
static __device__ __forceinline__ uint32_t lds_addr32(const void* p) {
  return (uint32_t)(uintptr_t)p;   // low 32 bits of flat addr == LDS offset
}
static __device__ __forceinline__ void async_b128(uint32_t lds, const void* sbase,
                                                  uint32_t voff) {
  asm volatile("global_load_async_to_lds_b128 %0, %1, %2"
               :: "v"(lds), "v"(voff), "s"(sbase)
               : "memory");
}
static __device__ __forceinline__ void wait_async0() {
  asm volatile("s_wait_asynccnt 0x0" ::: "memory");
}
static __device__ __forceinline__ void wait_async4() {
  asm volatile("s_wait_asynccnt 0x4" ::: "memory");
}
static __device__ __forceinline__ void wait_async8() {
  asm volatile("s_wait_asynccnt 0x8" ::: "memory");
}

// ---------------------------------------------------------------------------
// f32 -> bf16 convert (x)
// ---------------------------------------------------------------------------
__global__ __launch_bounds__(256) void cvt_f32_bf16(const float* __restrict__ src,
                                                    __bf16* __restrict__ dst, int n) {
  int i = blockIdx.x * blockDim.x + threadIdx.x;
  if (i < n) dst[i] = (__bf16)src[i];
}

// ---------------------------------------------------------------------------
// f32 [K][N] -> bf16 [N][K] transposing convert (weights; 32x32 LDS tiles)
// grid: (N/32, K/32), block 256
// ---------------------------------------------------------------------------
__global__ __launch_bounds__(256) void cvt_transpose_f32_bf16(const float* __restrict__ src,
                                                              __bf16* __restrict__ dst,
                                                              int K, int N) {
  __shared__ float tile[32][33];
  const int t  = threadIdx.x;
  const int bk = blockIdx.y * 32;
  const int bn = blockIdx.x * 32;
#pragma unroll
  for (int i = 0; i < 4; ++i) {
    const int idx = t + i * 256;
    const int r = idx >> 5, c = idx & 31;
    tile[r][c] = src[(size_t)(bk + r) * N + bn + c];
  }
  __syncthreads();
#pragma unroll
  for (int i = 0; i < 4; ++i) {
    const int idx = t + i * 256;
    const int r = idx >> 5, c = idx & 31;
    dst[(size_t)(bn + r) * K + bk + c] = (__bf16)tile[c][r];
  }
}

// ---------------------------------------------------------------------------
// Tiled bf16 WMMA GEMM:  C[M,N] = A[M,K] @ Bt[N,K]^T + bias[N]
// Bt pre-transposed (k contiguous) -> both tiles are straight async B128
// copies into LDS. Double-buffered k-steps (BK=32) via ASYNCcnt.
// 256 threads = 8 waves; block tile 128x128; wave tile 64x32.
// ---------------------------------------------------------------------------
template <bool OUT_BF16>
__global__ __launch_bounds__(256) void gemm_wmma_bf16(const __bf16* __restrict__ A,
                                                      const __bf16* __restrict__ Bt,
                                                      const float* __restrict__ bias,
                                                      void* __restrict__ Cout,
                                                      int M, int N, int K) {
  __shared__ __bf16 As[2][128][32];   // [m][k]
  __shared__ __bf16 Bs[2][128][32];   // [n][k]

  const int tid  = threadIdx.x;
  const int lane = tid & 31;
  const int wid  = tid >> 5;
  const int wrow = wid >> 2;          // 0..1
  const int wcol = wid & 3;           // 0..3
  const int l16  = lane & 15;
  const int lhi  = lane >> 4;
  const int kb   = lhi * 16;          // A-frag K base
  const int kh   = lhi * 16;          // B-frag K base

  const int bm = blockIdx.y * 128;
  const int bn = blockIdx.x * 128;

  const int ldrow = tid >> 1;           // 0..127
  const int ldcol = (tid & 1) * 16;     // 0 or 16 (elements)

  v8f acc[4][2] = {};

  auto issue_tile = [&](int buf, int k0) {
    const uint32_t aoff = (uint32_t)(((size_t)(bm + ldrow) * K + k0 + ldcol) * 2);
    async_b128(lds_addr32(&As[buf][ldrow][ldcol]),     A,  aoff);
    async_b128(lds_addr32(&As[buf][ldrow][ldcol + 8]), A,  aoff + 16);
    const uint32_t boff = (uint32_t)(((size_t)(bn + ldrow) * K + k0 + ldcol) * 2);
    async_b128(lds_addr32(&Bs[buf][ldrow][ldcol]),     Bt, boff);
    async_b128(lds_addr32(&Bs[buf][ldrow][ldcol + 8]), Bt, boff + 16);
  };

  const int nk = K >> 5;
  issue_tile(0, 0);
  for (int kt = 0; kt < nk; ++kt) {
    const int buf = kt & 1;
    if (kt + 1 < nk) {
      issue_tile(buf ^ 1, (kt + 1) * 32);
      wait_async4();                    // first 4 (current tile) complete in order
    } else {
      wait_async0();
    }
    __syncthreads();

    v16bf af[4], bf[2];
#pragma unroll
    for (int i = 0; i < 4; ++i) {
      const int row = wrow * 64 + i * 16 + l16;
      af[i] = load_frag_halves(&As[buf][row][kb], &As[buf][row][16 + kb]);
    }
#pragma unroll
    for (int j = 0; j < 2; ++j) {
      const int n = wcol * 32 + j * 16 + l16;
      bf[j] = load_frag_contig(&Bs[buf][n][kh]);
    }
#pragma unroll
    for (int i = 0; i < 4; ++i)
#pragma unroll
      for (int j = 0; j < 2; ++j)
        acc[i][j] = __builtin_amdgcn_wmma_f32_16x16x32_bf16(
            false, af[i], false, bf[j], (short)0, acc[i][j], false, false);

    __syncthreads();   // compute done before this buffer is refilled
  }

#pragma unroll
  for (int i = 0; i < 4; ++i) {
#pragma unroll
    for (int j = 0; j < 2; ++j) {
      const int col = bn + wcol * 32 + j * 16 + l16;
      const float bv = bias[col];
#pragma unroll
      for (int v = 0; v < 8; ++v) {
        const int row = bm + wrow * 64 + i * 16 + v + lhi * 8;
        const float val = acc[i][j][v] + bv;
        if (OUT_BF16)
          reinterpret_cast<__bf16*>(Cout)[(size_t)row * N + col] = (__bf16)val;
        else
          reinterpret_cast<float*>(Cout)[(size_t)row * N + col] = val;
      }
    }
  }
}

// ---------------------------------------------------------------------------
// V transpose: qkv[b*T+t][2C + h*64 + d] -> vT[((b*H+h)*64 + d)*T + t]
// One block per (b,h, 64-row tile). 128 threads.
// ---------------------------------------------------------------------------
__global__ __launch_bounds__(128) void transpose_v(const __bf16* __restrict__ qkv,
                                                   __bf16* __restrict__ vT) {
  constexpr int T = 2048, C = 1024, H = 16;
  __shared__ __bf16 tile[64][72];   // pad keeps 16B alignment of [r][0]/[r][32]
  const int tid = threadIdx.x;
  const int b = blockIdx.y / H, h = blockIdx.y % H;
  const int t0 = blockIdx.x * 64;
  const int r   = tid >> 1;
  const int off = (tid & 1) * 32;
  {
    const __bf16* src = qkv + (size_t)(b * T + t0 + r) * (3 * C) + 2 * C + h * 64 + off;
    const uint4* s4 = reinterpret_cast<const uint4*>(src);
    uint4* d4 = reinterpret_cast<uint4*>(&tile[r][off]);
    d4[0] = s4[0]; d4[1] = s4[1]; d4[2] = s4[2]; d4[3] = s4[3];
  }
  __syncthreads();
  {
    __bf16 tv[32];
#pragma unroll
    for (int i = 0; i < 32; ++i) tv[i] = tile[off + i][r];
    __bf16* dst = vT + ((size_t)(b * H + h) * 64 + r) * T + t0 + off;
    uint4* d4 = reinterpret_cast<uint4*>(dst);
    const uint4* s4 = reinterpret_cast<const uint4*>(&tv[0]);
    d4[0] = s4[0]; d4[1] = s4[1]; d4[2] = s4[2]; d4[3] = s4[3];
  }
}

// ---------------------------------------------------------------------------
// Flash attention (causal, online softmax). One block = 64 q rows of one (b,h).
// 128 threads = 4 waves; wave w owns q rows [qbase+16w, qbase+16w+16).
// K/V tiles double-buffered via async B128 -> LDS; diagonal tile peeled so the
// steady-state loop carries no mask VALU.
// ---------------------------------------------------------------------------
__global__ __launch_bounds__(128) void flash_attn_wmma(const __bf16* __restrict__ qkv,
                                                       const __bf16* __restrict__ vT,
                                                       __bf16* __restrict__ y) {
  constexpr int T = 2048, C = 1024, H = 16, D = 64;
  constexpr float SCALE = 0.125f;  // 1/sqrt(64)

  __shared__ __bf16 Ks[2][64][64];     // [t_local][d]
  __shared__ __bf16 Vs[2][64][64];     // [d][t_local]
  __shared__ __bf16 Ps[4][16][64];     // per-wave P tile, [row][key]

  const int tid  = threadIdx.x;
  const int lane = tid & 31;
  const int wid  = tid >> 5;
  const int l16  = lane & 15;
  const int lhi  = lane >> 4;
  const int kb   = lhi * 16;
  const int kh   = lhi * 16;

  const int bh = blockIdx.y;
  const int b  = bh / H;
  const int h  = bh % H;
  const int qtile = blockIdx.x;
  const int qbase = qtile * 64;
  const size_t rstride = 3 * C;

  // Q fragments (A operand), loaded once
  v16bf aq[2];
  {
    const int t = qbase + wid * 16 + l16;
    const __bf16* qrow = qkv + (size_t)(b * T + t) * rstride + h * D;
#pragma unroll
    for (int s = 0; s < 2; ++s)
      aq[s] = load_frag_halves(qrow + s * 32 + kb, qrow + s * 32 + 16 + kb);
  }

  float m[8], l[8], alpha[8];
  v8f o[4] = {};
#pragma unroll
  for (int v = 0; v < 8; ++v) { m[v] = -3.0e38f; l[v] = 0.0f; }

  const int ldr = tid >> 1;          // 0..63
  const int ldo = (tid & 1) * 32;    // 0 or 32 (elements), 64B per thread

  // issue one K/V tile pair (8 async B128 per thread)
  auto issue_tiles = [&](int buf, int kt) {
    const uint32_t koff =
        (uint32_t)(((size_t)(b * T + kt * 64 + ldr) * rstride + C + h * D + ldo) * 2);
#pragma unroll
    for (int i = 0; i < 4; ++i)
      async_b128(lds_addr32(&Ks[buf][ldr][ldo + 8 * i]), qkv, koff + 16 * i);
    const uint32_t voff =
        (uint32_t)((((size_t)(b * H + h) * 64 + ldr) * T + kt * 64 + ldo) * 2);
#pragma unroll
    for (int i = 0; i < 4; ++i)
      async_b128(lds_addr32(&Vs[buf][ldr][ldo + 8 * i]), vT, voff + 16 * i);
  };

  // one k-tile of the flash recurrence; DIAG selects causal masking
  auto process_tile = [&](int buf, int kt, bool diag) {
    // S = q @ k^T (16x64 per wave)
    v8f sacc[4] = {};
#pragma unroll
    for (int s = 0; s < 2; ++s) {
#pragma unroll
      for (int j = 0; j < 4; ++j) {
        v16bf bk = load_frag_contig(&Ks[buf][j * 16 + l16][s * 32 + kh]);
        sacc[j] = __builtin_amdgcn_wmma_f32_16x16x32_bf16(
            false, aq[s], false, bk, (short)0, sacc[j], false, false);
      }
    }

    float sv[4][8];
    if (diag) {
      const int qg = qbase + wid * 16 + lhi * 8;   // + v below
#pragma unroll
      for (int j = 0; j < 4; ++j) {
        const int kg = kt * 64 + j * 16 + l16;
#pragma unroll
        for (int v = 0; v < 8; ++v) {
          float x = sacc[j][v] * SCALE;
          if (kg > qg + v) x = -3.0e38f;
          sv[j][v] = x;
        }
      }
    } else {
#pragma unroll
      for (int j = 0; j < 4; ++j)
#pragma unroll
        for (int v = 0; v < 8; ++v)
          sv[j][v] = sacc[j][v] * SCALE;
    }

    // online softmax
#pragma unroll
    for (int v = 0; v < 8; ++v) {
      float mt = fmaxf(fmaxf(sv[0][v], sv[1][v]), fmaxf(sv[2][v], sv[3][v]));
      mt = half16_max(mt);
      const float mnew = fmaxf(m[v], mt);
      alpha[v] = __expf(m[v] - mnew);
      m[v] = mnew;
    }
#pragma unroll
    for (int j = 0; j < 4; ++j)
#pragma unroll
      for (int v = 0; v < 8; ++v)
        sv[j][v] = __expf(sv[j][v] - m[v]);
#pragma unroll
    for (int v = 0; v < 8; ++v) {
      float rs = sv[0][v] + sv[1][v] + sv[2][v] + sv[3][v];
      rs = half16_sum(rs);
      l[v] = l[v] * alpha[v] + rs;
    }
#pragma unroll
    for (int j = 0; j < 4; ++j)
#pragma unroll
      for (int v = 0; v < 8; ++v)
        o[j][v] *= alpha[v];

    // restripe P (C-layout) into A-fragment layout via per-wave LDS region
#pragma unroll
    for (int j = 0; j < 4; ++j)
#pragma unroll
      for (int v = 0; v < 8; ++v)
        Ps[wid][v + lhi * 8][j * 16 + l16] = (__bf16)sv[j][v];

    // O += P @ V
#pragma unroll
    for (int s = 0; s < 2; ++s) {
      v16bf ap = load_frag_halves(&Ps[wid][l16][s * 32 + kb],
                                  &Ps[wid][l16][s * 32 + 16 + kb]);
#pragma unroll
      for (int j = 0; j < 4; ++j) {
        v16bf bv = load_frag_contig(&Vs[buf][j * 16 + l16][s * 32 + kh]);
        o[j] = __builtin_amdgcn_wmma_f32_16x16x32_bf16(
            false, ap, false, bv, (short)0, o[j], false, false);
      }
    }
  };

  issue_tiles(0, 0);
  // steady-state: tiles 0..qtile-1, no masking, next tile prefetched async
  for (int kt = 0; kt < qtile; ++kt) {
    const int buf = kt & 1;
    issue_tiles(buf ^ 1, kt + 1);
    wait_async8();                    // current tile's 8 done (in-order)
    __syncthreads();
    process_tile(buf, kt, false);
    __syncthreads();                  // before buf is refilled next iteration
  }
  // peeled diagonal tile (kt == qtile)
  {
    const int buf = qtile & 1;
    wait_async0();
    __syncthreads();
    process_tile(buf, qtile, true);
  }

  // normalize and store y (bf16)
#pragma unroll
  for (int j = 0; j < 4; ++j)
#pragma unroll
    for (int v = 0; v < 8; ++v) {
      const int row = qbase + wid * 16 + v + lhi * 8;
      const int d   = j * 16 + l16;
      const float val = o[j][v] / l[v];
      y[(size_t)(b * T + row) * C + h * D + d] = (__bf16)val;
    }
}

// ---------------------------------------------------------------------------
// Launcher
// ---------------------------------------------------------------------------
extern "C" void kernel_launch(void* const* d_in, const int* in_sizes, int n_in,
                              void* d_out, int out_size, void* d_ws, size_t ws_size,
                              hipStream_t stream) {
  constexpr int B = 4, T = 2048, C = 1024, H = 16;
  constexpr int M = B * T;        // 8192
  constexpr int N1 = 3 * C;       // 3072
  constexpr int N2 = C;           // 1024

  const float* x      = (const float*)d_in[0];
  const float* w_attn = (const float*)d_in[1];
  const float* b_attn = (const float*)d_in[2];
  const float* w_proj = (const float*)d_in[3];
  const float* b_proj = (const float*)d_in[4];

  char* ws = (char*)d_ws;
  size_t off = 0;
  auto carve = [&](size_t bytes) {
    char* p = ws + off;
    off = (off + bytes + 255) & ~(size_t)255;
    return p;
  };
  __bf16* xb    = (__bf16*)carve((size_t)M * C * 2);    // x bf16 [M][C]
  __bf16* wabT  = (__bf16*)carve((size_t)N1 * C * 2);   // w_attn^T bf16 [3C][C]
  __bf16* wpbT  = (__bf16*)carve((size_t)N2 * C * 2);   // w_proj^T bf16 [C][C]
  __bf16* qkvb  = (__bf16*)carve((size_t)M * N1 * 2);   // qkv bf16 [M][3C]
  __bf16* vTb   = (__bf16*)carve((size_t)B * H * 64 * T * 2); // V^T [b,h,d][t]
  __bf16* yb    = (__bf16*)carve((size_t)M * C * 2);    // attn out bf16 [M][C]
  (void)ws_size;

  // 1) converts (x straight; weights transposed to [N][K])
  {
    int n = M * C;
    cvt_f32_bf16<<<(n + 255) / 256, 256, 0, stream>>>(x, xb, n);
    cvt_transpose_f32_bf16<<<dim3(N1 / 32, C / 32), 256, 0, stream>>>(w_attn, wabT, C, N1);
    cvt_transpose_f32_bf16<<<dim3(N2 / 32, C / 32), 256, 0, stream>>>(w_proj, wpbT, C, N2);
  }

  // 2) qkv = x @ w_attn + b_attn   (bf16 out)
  gemm_wmma_bf16<true><<<dim3(N1 / 128, M / 128), 256, 0, stream>>>(
      xb, wabT, b_attn, qkvb, M, N1, C);

  // 3) V transpose for t-contiguous B-fragments in flash
  transpose_v<<<dim3(T / 64, B * H), 128, 0, stream>>>(qkvb, vTb);

  // 4) flash attention -> yb (bf16)
  flash_attn_wmma<<<dim3(T / 64, B * H), 128, 0, stream>>>(qkvb, vTb, yb);

  // 5) out = y @ w_proj + b_proj   (f32 out)
  gemm_wmma_bf16<false><<<dim3(N2 / 128, M / 128), 256, 0, stream>>>(
      yb, wpbT, b_proj, (float*)d_out, M, N2, C);
}
